// MultiheadAttention_21500606283919
// MI455X (gfx1250) — matrix operand order repaired and checked by reading, not compiled, not run on hardware
//
#include <hip/hip_runtime.h>
#include <stdint.h>

// Problem constants (from reference): B=4, S=2048, D=1024, H=16, Hd=64
#define BDIM 4
#define SEQ  2048
#define EMB  1024
#define NH   16
#define HD   64

typedef __attribute__((ext_vector_type(16))) _Float16 v16h;
typedef __attribute__((ext_vector_type(8)))  _Float16 v8h;
typedef __attribute__((ext_vector_type(8)))  float    v8f;

// ---------------------------------------------------------------------------
// WMMA fragment loaders (wave32), this code's contiguous k order (same for A and B):
//  A (16x32 f16): lane l -> row M=l&15; halves 0..15 = K k0..k0+15,
//                 k0 = 16*(l>>4)
//  B (32x16 f16): lane l -> col N=l&15; halves 0..15 = K k0..k0+15,
//                 k0 = 16*(l>>4)
//  C (16x16 f32): VGPR r -> row M = r + 8*(l>>4), col N = l&15
// ---------------------------------------------------------------------------
static __device__ __forceinline__
v16h load_a_frag(const _Float16* base, int ld, int mr, int kb, int lane) {
  int m  = mr + (lane & 15);
  int k0 = kb + ((lane >> 4) << 4);
  return *(const v16h*)(base + (size_t)m * ld + k0);
}

static __device__ __forceinline__
v16h load_b_frag(const _Float16* base, int ld, int nb, int kb, int lane) {
  int n  = nb + (lane & 15);
  int k0 = kb + ((lane >> 4) << 4);
  return *(const v16h*)(base + (size_t)n * ld + k0);
}

static __device__ __forceinline__ v8f wmma16(v16h a, v16h b, v8f c) {
  return __builtin_amdgcn_wmma_f32_16x16x32_f16(false, a, false, b,
                                                (short)0, c, false, false);
}

// Generic (flat) pointer to a __shared__ object -> raw 32-bit LDS byte offset,
// as required by the async-to-LDS instructions' VDST operand.
static __device__ __forceinline__ unsigned lds_offset(const void* p) {
  return (unsigned)(size_t)(__attribute__((address_space(3))) const void*)p;
}

// ---------------------------------------------------------------------------
// f32 -> f16 down-convert (grid-stride)
// ---------------------------------------------------------------------------
__global__ void cvt_f32_f16(const float* __restrict__ in,
                            _Float16* __restrict__ out, int n) {
  int stride = gridDim.x * blockDim.x;
  for (int i = blockIdx.x * blockDim.x + threadIdx.x; i < n; i += stride)
    out[i] = (_Float16)in[i];
}

// ---------------------------------------------------------------------------
// V [B,S,D] -> Vt [B,H,Hd,S]  (so PV WMMA B-operand is contiguous along j)
// ---------------------------------------------------------------------------
__global__ void vtranspose(const _Float16* __restrict__ V,
                           _Float16* __restrict__ Vt) {
  const int total = BDIM * NH * HD * SEQ;  // 2^23
  int stride = gridDim.x * blockDim.x;
  for (int t = blockIdx.x * blockDim.x + threadIdx.x; t < total; t += stride) {
    int j = t & (SEQ - 1);
    int d = (t >> 11) & (HD - 1);
    int h = (t >> 17) & (NH - 1);
    int b = t >> 21;
    Vt[t] = V[((size_t)(b * SEQ + j)) * EMB + h * HD + d];
  }
}

// ---------------------------------------------------------------------------
// Y[M,N] = X[M,K] @ W[N,K]^T   (nn.Linear convention: x @ W.T)
// 4 waves/block, wave -> 64x64 tile (4x4 accumulator frags), block -> 128x128.
// Weights (2MB) are fully L2-resident; B-frags are register-reused 4x, so
// direct global loads are sufficient here.
// ---------------------------------------------------------------------------
template <bool F32OUT>
__global__ __launch_bounds__(128)
void gemm_xwT(const _Float16* __restrict__ X, const _Float16* __restrict__ W,
              _Float16* __restrict__ Yh, float* __restrict__ Yf,
              const float* __restrict__ bias, int M, int N, int Kd) {
  const int lane = threadIdx.x & 31;
  const int wave = threadIdx.x >> 5;
  const int m0 = blockIdx.y * 128 + (wave >> 1) * 64;
  const int n0 = blockIdx.x * 128 + (wave & 1) * 64;

  v8f c[4][4] = {};
  for (int kb = 0; kb < Kd; kb += 32) {
    v16h a[4];
#pragma unroll
    for (int mt = 0; mt < 4; ++mt)
      a[mt] = load_a_frag(X, Kd, m0 + mt * 16, kb, lane);
#pragma unroll
    for (int nt = 0; nt < 4; ++nt) {
      v16h bf = load_b_frag(W, Kd, n0 + nt * 16, kb, lane);
#pragma unroll
      for (int mt = 0; mt < 4; ++mt) c[mt][nt] = wmma16(a[mt], bf, c[mt][nt]);
    }
  }

  const int lo = lane & 15, hi = lane >> 4;
#pragma unroll
  for (int mt = 0; mt < 4; ++mt)
#pragma unroll
    for (int nt = 0; nt < 4; ++nt)
#pragma unroll
      for (int r = 0; r < 8; ++r) {
        int m = m0 + mt * 16 + r + 8 * hi;
        int n = n0 + nt * 16 + lo;
        size_t idx = (size_t)m * N + n;
        if constexpr (F32OUT) Yf[idx] = c[mt][nt][r] + bias[n];
        else                  Yh[idx] = (_Float16)c[mt][nt][r];
      }
}

// ---------------------------------------------------------------------------
// Flash attention. 1 wave = 32 query rows x Hd=64; 4 waves/block = 128 rows.
// All 4 waves share (b,h) => K/V tiles for each 32-key block are staged ONCE
// per block into LDS with GLOBAL_LOAD_ASYNC_TO_LDS_B128 (ASYNCcnt), then read
// back as WMMA B-fragments via ds loads. Score C-layout -> P A-layout
// transpose goes through a per-wave LDS tile.
// LDS: kt 4KB + vt 4KB + plds 8KB = 16KB/block.
// ---------------------------------------------------------------------------
__global__ __launch_bounds__(128)
void flash_attn(const _Float16* __restrict__ Q, const _Float16* __restrict__ K,
                const _Float16* __restrict__ Vt, _Float16* __restrict__ C) {
  __shared__ _Float16 kt[32][64];        // K tile: 32 keys x Hd
  __shared__ _Float16 vt[64][32];        // Vt tile: Hd x 32 keys
  __shared__ _Float16 plds[4][32][32];   // per-wave P transpose staging

  const int tid  = threadIdx.x;
  const int lane = tid & 31;
  const int wave = tid >> 5;
  const int lo = lane & 15, hi = lane >> 4;
  const int bh = blockIdx.y;              // b*NH + h
  const int b = bh >> 4, h = bh & 15;
  const int q0 = blockIdx.x * 128 + wave * 32;

  const _Float16* Qbase = Q + ((size_t)(b * SEQ + q0)) * EMB + h * HD;
  const _Float16* Kbase = K + ((size_t)b * SEQ) * EMB + h * HD;
  const _Float16* Vbase = Vt + ((size_t)bh * HD) * SEQ;   // [Hd][S]

  // Preload Q fragments: 2 M-tiles x (Hd=64 -> 2 K-steps of 32)
  v16h qa[2][2];
#pragma unroll
  for (int mt = 0; mt < 2; ++mt)
#pragma unroll
    for (int ks = 0; ks < 2; ++ks)
      qa[mt][ks] = load_a_frag(Qbase, EMB, mt * 16, ks * 32, lane);

  v8f acc[2][4] = {};
  float mstat[2][8], lstat[2][8];
#pragma unroll
  for (int mt = 0; mt < 2; ++mt)
#pragma unroll
    for (int r = 0; r < 8; ++r) { mstat[mt][r] = -3.0e38f; lstat[mt][r] = 0.f; }

  const float scale = 0.125f;  // 1/sqrt(Hd)

  for (int jb = 0; jb < SEQ; jb += 32) {
    // Previous iteration's LDS tile reads are all consumed by its WMMAs, so a
    // single barrier protects against overwriting tiles another wave still
    // needs, then async-stage the new K/V tiles cooperatively.
    __syncthreads();
#pragma unroll
    for (int i = 0; i < 2; ++i) {
      int c = tid + i * 128;                 // 0..255 chunk id (16B chunks)
      // K tile: 32 rows x 128B  (8 chunks per row)
      int kr = c >> 3, kc = (c & 7) * 8;
      const _Float16* gk = Kbase + (size_t)(jb + kr) * EMB + kc;
      unsigned lk = lds_offset(&kt[kr][kc]);
      asm volatile("global_load_async_to_lds_b128 %0, %1, off"
                   :: "v"(lk), "v"(gk) : "memory");
      // V tile: 64 rows x 64B  (4 chunks per row)
      int vr = c >> 2, vc = (c & 3) * 8;
      const _Float16* gv = Vbase + (size_t)vr * SEQ + jb + vc;
      unsigned lv = lds_offset(&vt[vr][vc]);
      asm volatile("global_load_async_to_lds_b128 %0, %1, off"
                   :: "v"(lv), "v"(gv) : "memory");
    }
    // Prefetch next block's K/V rows into cache while we compute.
    if (jb + 32 < SEQ) {
      __builtin_prefetch(Kbase + (size_t)(jb + 32 + lane) * EMB, 0, 1);
      __builtin_prefetch(Vbase + (size_t)(2 * lane) * SEQ + jb + 32, 0, 1);
    }
    asm volatile("s_wait_asynccnt 0x0" ::: "memory");
    __syncthreads();

    // ---- scores S = Q K^T for the 32-key block (B-frags from LDS) ----
    v8f s[2][2] = {};
#pragma unroll
    for (int nt = 0; nt < 2; ++nt)
#pragma unroll
      for (int ks = 0; ks < 2; ++ks) {
        v16h kf = load_b_frag(&kt[0][0], 64, nt * 16, ks * 32, lane);
#pragma unroll
        for (int mt = 0; mt < 2; ++mt) s[mt][nt] = wmma16(qa[mt][ks], kf, s[mt][nt]);
      }

    // ---- online softmax (row = r + 8*hi; 16-lane xor reductions) ----
#pragma unroll
    for (int mt = 0; mt < 2; ++mt) {
#pragma unroll
      for (int r = 0; r < 8; ++r) {
        float s0 = s[mt][0][r] * scale;
        float s1 = s[mt][1][r] * scale;
        float t = fmaxf(s0, s1);
        t = fmaxf(t, __shfl_xor(t, 1, 32));
        t = fmaxf(t, __shfl_xor(t, 2, 32));
        t = fmaxf(t, __shfl_xor(t, 4, 32));
        t = fmaxf(t, __shfl_xor(t, 8, 32));
        float mnew  = fmaxf(mstat[mt][r], t);
        float alpha = __expf(mstat[mt][r] - mnew);
        mstat[mt][r] = mnew;
        float p0 = __expf(s0 - mnew);
        float p1 = __expf(s1 - mnew);
        int row = mt * 16 + r + 8 * hi;
        plds[wave][row][lo]      = (_Float16)p0;   // C-layout -> LDS
        plds[wave][row][16 + lo] = (_Float16)p1;
        float u = p0 + p1;
        u += __shfl_xor(u, 1, 32);
        u += __shfl_xor(u, 2, 32);
        u += __shfl_xor(u, 4, 32);
        u += __shfl_xor(u, 8, 32);
        lstat[mt][r] = alpha * lstat[mt][r] + u;
#pragma unroll
        for (int nt = 0; nt < 4; ++nt) acc[mt][nt][r] *= alpha;
      }
    }

    // ---- ctx += P V : P re-read from LDS in A-layout, V-frags from LDS ----
    v16h pa[2];
#pragma unroll
    for (int mt = 0; mt < 2; ++mt)
      pa[mt] = load_a_frag(&plds[wave][0][0], 32, mt * 16, 0, lane);
#pragma unroll
    for (int nt = 0; nt < 4; ++nt) {
      v16h vf = load_b_frag(&vt[0][0], 32, nt * 16, 0, lane);
#pragma unroll
      for (int mt = 0; mt < 2; ++mt) acc[mt][nt] = wmma16(pa[mt], vf, acc[mt][nt]);
    }
  }

  // ---- normalize and write ctx (f16, [B,S,D] layout) ----
#pragma unroll
  for (int mt = 0; mt < 2; ++mt)
#pragma unroll
    for (int r = 0; r < 8; ++r) {
      float inv = 1.0f / lstat[mt][r];
      int m = q0 + mt * 16 + r + 8 * hi;
#pragma unroll
      for (int nt = 0; nt < 4; ++nt) {
        int d = h * HD + nt * 16 + lo;
        C[((size_t)(b * SEQ + m)) * EMB + d] = (_Float16)(acc[mt][nt][r] * inv);
      }
    }
}

// ---------------------------------------------------------------------------
// Launch pipeline. Workspace layout (f16), ~104 MB total:
//   Xh(16M) | Wq/Wk/Wv/Wo(4x2M) | Qh(16M) | Kh(16M) | Vh(16M) | Vth(16M) | Ch(16M)
// ---------------------------------------------------------------------------
extern "C" void kernel_launch(void* const* d_in, const int* in_sizes, int n_in,
                              void* d_out, int out_size, void* d_ws, size_t ws_size,
                              hipStream_t stream) {
  (void)in_sizes; (void)n_in; (void)out_size; (void)ws_size;
  const float* x  = (const float*)d_in[0];
  const float* wq = (const float*)d_in[1];
  const float* wk = (const float*)d_in[2];
  const float* wv = (const float*)d_in[3];
  const float* wo = (const float*)d_in[4];
  const float* bo = (const float*)d_in[5];
  float* out = (float*)d_out;

  char* ws = (char*)d_ws;
  const size_t sz_x = (size_t)BDIM * SEQ * EMB * sizeof(_Float16);  // 16 MB
  const size_t sz_w = (size_t)EMB * EMB * sizeof(_Float16);         //  2 MB
  _Float16* Xh  = (_Float16*)(ws);
  _Float16* Wqh = (_Float16*)(ws + sz_x);
  _Float16* Wkh = (_Float16*)(ws + sz_x + 1 * sz_w);
  _Float16* Wvh = (_Float16*)(ws + sz_x + 2 * sz_w);
  _Float16* Woh = (_Float16*)(ws + sz_x + 3 * sz_w);
  _Float16* Qh  = (_Float16*)(ws + 1 * sz_x + 4 * sz_w);
  _Float16* Kh  = (_Float16*)(ws + 2 * sz_x + 4 * sz_w);
  _Float16* Vh  = (_Float16*)(ws + 3 * sz_x + 4 * sz_w);
  _Float16* Vth = (_Float16*)(ws + 4 * sz_x + 4 * sz_w);
  _Float16* Ch  = (_Float16*)(ws + 5 * sz_x + 4 * sz_w);

  const int M = BDIM * SEQ;   // 8192
  const int N = EMB;          // 1024
  const int Kd = EMB;         // 1024

  cvt_f32_f16<<<512, 256, 0, stream>>>(x,  Xh,  M * Kd);
  cvt_f32_f16<<<256, 256, 0, stream>>>(wq, Wqh, N * Kd);
  cvt_f32_f16<<<256, 256, 0, stream>>>(wk, Wkh, N * Kd);
  cvt_f32_f16<<<256, 256, 0, stream>>>(wv, Wvh, N * Kd);
  cvt_f32_f16<<<256, 256, 0, stream>>>(wo, Woh, N * Kd);

  dim3 ggrid(N / 128, M / 128);  // (8, 64)
  gemm_xwT<false><<<ggrid, 128, 0, stream>>>(Xh, Wqh, Qh, nullptr, nullptr, M, N, Kd);
  gemm_xwT<false><<<ggrid, 128, 0, stream>>>(Xh, Wkh, Kh, nullptr, nullptr, M, N, Kd);
  gemm_xwT<false><<<ggrid, 128, 0, stream>>>(Xh, Wvh, Vh, nullptr, nullptr, M, N, Kd);

  vtranspose<<<1024, 256, 0, stream>>>(Vh, Vth);

  flash_attn<<<dim3(SEQ / 128, BDIM * NH), 128, 0, stream>>>(Qh, Kh, Vth, Ch);

  gemm_xwT<true><<<ggrid, 128, 0, stream>>>(Ch, Woh, nullptr, out, bo, M, N, Kd);
}
